// QLoRABigNet_53798760350388
// MI455X (gfx1250) — matrix operand order repaired and checked by reading, not compiled, not run on hardware
//
#include <hip/hip_runtime.h>
#include <hip/hip_bf16.h>

#define CD   1024          // BIGNET_DIM
#define NB   4096          // batch
#define NGRP 8192          // groups per layer (C*C/GS)
#define GS   128

typedef __attribute__((ext_vector_type(16))) _Float16 v16h;
typedef __attribute__((ext_vector_type(8)))  _Float16 v8h;
typedef __attribute__((ext_vector_type(8)))  float    v8f;

union F16Frag { v16h v; v8h h[2]; };

// ---------------------------------------------------------------------------
// Dequant 4-bit weights + fold LoRA rank-4 update: W_eff[n][k] = W[n][k] + b[n]·a[:,k]
// One thread per packed int32 (two nibbles -> two consecutive k).
// ---------------------------------------------------------------------------
__global__ __launch_bounds__(256)
void dequant_kernel(const int* __restrict__ q4, const float* __restrict__ norm,
                    const float* __restrict__ la,   // [4][CD]
                    const float* __restrict__ lb,   // [CD][4]
                    _Float16* __restrict__ wout)    // [CD][CD] row-major (k contiguous)
{
    int t = blockIdx.x * blockDim.x + threadIdx.x;
    if (t >= CD * CD / 2) return;
    int g = t >> 6;                       // group index (64 ints per group)
    int j = t & 63;                       // int within group
    int n = g >> 3;                       // output row (8 groups of 128 per row)
    int k = ((g & 7) << 7) + (j << 1);    // column base

    int   q = q4[t];
    float s = norm[g];
    float w0 = ((float)( q       & 0xF) * (2.0f / 15.0f) - 1.0f) * s;
    float w1 = ((float)((q >> 4) & 0xF) * (2.0f / 15.0f) - 1.0f) * s;

    const float* bn = lb + n * 4;
    float b0 = bn[0], b1 = bn[1], b2 = bn[2], b3 = bn[3];
    w0 += b0 * la[0*CD + k]     + b1 * la[1*CD + k]     + b2 * la[2*CD + k]     + b3 * la[3*CD + k];
    w1 += b0 * la[0*CD + k + 1] + b1 * la[1*CD + k + 1] + b2 * la[2*CD + k + 1] + b3 * la[3*CD + k + 1];

    size_t o = (size_t)n * CD + k;
    wout[o]     = (_Float16)w0;
    wout[o + 1] = (_Float16)w1;
}

// ---------------------------------------------------------------------------
// f32 -> f16 convert
// ---------------------------------------------------------------------------
__global__ __launch_bounds__(256)
void cvt_kernel(const float* __restrict__ in, _Float16* __restrict__ out, int n)
{
    int t = blockIdx.x * blockDim.x + threadIdx.x;
    if (t < n) out[t] = (_Float16)in[t];
}

// ---------------------------------------------------------------------------
// GEMM: out[m][n] = sum_k act[m][k] * W[n][k]  (+relu | +resid)
// 256 threads = 8 waves. Block tile 128x128, wave tile 32x64 (2x4 frags of 16x16).
// v_wmma_f32_16x16x32_f16, K-loop step 32.
// ---------------------------------------------------------------------------
__global__ __launch_bounds__(256)
void gemm_kernel(const _Float16* __restrict__ A,   // [NB][CD] f16
                 const _Float16* __restrict__ W,   // [CD][CD] f16 (k contiguous)
                 const float* __restrict__ resid,  // [NB][CD] f32 or null
                 float* __restrict__ out32,        // [NB][CD] f32 or null
                 _Float16* __restrict__ out16,     // [NB][CD] f16
                 int relu)
{
    const int lane  = threadIdx.x & 31;
    const int wave  = threadIdx.x >> 5;
    const int waveM = wave & 3;     // 0..3
    const int waveN = wave >> 2;    // 0..1
    const int lmod  = lane & 15;
    const int lhalf = lane >> 4;

    const int m0 = blockIdx.y * 128 + waveM * 32;
    const int n0 = blockIdx.x * 128 + waveN * 64;

    v8f acc[2][4];
#pragma unroll
    for (int i = 0; i < 2; ++i)
#pragma unroll
        for (int jj = 0; jj < 4; ++jj)
#pragma unroll
            for (int e = 0; e < 8; ++e) acc[i][jj][e] = 0.0f;

    for (int kk = 0; kk < CD; kk += 32) {
        F16Frag a[2], b[4];
#pragma unroll
        for (int i = 0; i < 2; ++i) {
            // A 16x32 layout: lane holds row m0+i*16+lmod, 8 halfs at k+lhalf*8 and k+lhalf*8+16
            const _Float16* p = A + (size_t)(m0 + i * 16 + lmod) * CD + kk + lhalf * 8;
            a[i].h[0] = *(const v8h*)(p);
            a[i].h[1] = *(const v8h*)(p + 16);
        }
#pragma unroll
        for (int jj = 0; jj < 4; ++jj) {
            // B 32x16 layout: lane holds col n0+jj*16+lmod, 8 halfs at k+lhalf*8 and k+lhalf*8+16
            const _Float16* p = W + (size_t)(n0 + jj * 16 + lmod) * CD + kk + lhalf * 8;
            b[jj].h[0] = *(const v8h*)(p);
            b[jj].h[1] = *(const v8h*)(p + 16);
        }
#pragma unroll
        for (int i = 0; i < 2; ++i)
#pragma unroll
            for (int jj = 0; jj < 4; ++jj)
                acc[i][jj] = __builtin_amdgcn_wmma_f32_16x16x32_f16(
                    false, a[i].v, false, b[jj].v, (short)0, acc[i][jj], false, false);
    }

    // Epilogue: C/D layout -> VGPR r, lane L: m = m0+i*16 + lhalf*8 + r, n = n0+jj*16+lmod
#pragma unroll
    for (int i = 0; i < 2; ++i)
#pragma unroll
        for (int jj = 0; jj < 4; ++jj) {
            int n = n0 + jj * 16 + lmod;
#pragma unroll
            for (int r = 0; r < 8; ++r) {
                int m = m0 + i * 16 + lhalf * 8 + r;
                float v = acc[i][jj][r];
                if (relu) v = v > 0.0f ? v : 0.0f;
                size_t o = (size_t)m * CD + n;
                if (resid) v += resid[o];
                if (out32) out32[o] = v;
                out16[o] = (_Float16)v;
            }
        }
}

// ---------------------------------------------------------------------------
// LayerNorm over rows of C=1024; one 256-thread block per row. In-place f32 +
// writes f16 copy for next GEMM input.
// ---------------------------------------------------------------------------
__global__ __launch_bounds__(256)
void ln_kernel(float* __restrict__ h, _Float16* __restrict__ h16,
               const float* __restrict__ g, const float* __restrict__ b)
{
    __shared__ float red[256];
    const int row = blockIdx.x;
    float* p = h + (size_t)row * CD;

    float x[4];
    float s = 0.0f;
#pragma unroll
    for (int i = 0; i < 4; ++i) { x[i] = p[threadIdx.x + i * 256]; s += x[i]; }
    red[threadIdx.x] = s;
    __syncthreads();
    for (int off = 128; off > 0; off >>= 1) {
        if (threadIdx.x < off) red[threadIdx.x] += red[threadIdx.x + off];
        __syncthreads();
    }
    float mu = red[0] * (1.0f / CD);
    __syncthreads();

    float v = 0.0f;
#pragma unroll
    for (int i = 0; i < 4; ++i) { float d = x[i] - mu; v += d * d; }
    red[threadIdx.x] = v;
    __syncthreads();
    for (int off = 128; off > 0; off >>= 1) {
        if (threadIdx.x < off) red[threadIdx.x] += red[threadIdx.x + off];
        __syncthreads();
    }
    float inv = rsqrtf(red[0] * (1.0f / CD) + 1e-5f);
    __syncthreads();

#pragma unroll
    for (int i = 0; i < 4; ++i) {
        int c = threadIdx.x + i * 256;
        float y = (x[i] - mu) * inv * g[c] + b[c];
        p[c] = y;
        h16[(size_t)row * CD + c] = (_Float16)y;
    }
}

// ---------------------------------------------------------------------------
// Host orchestration
// ---------------------------------------------------------------------------
extern "C" void kernel_launch(void* const* d_in, const int* in_sizes, int n_in,
                              void* d_out, int out_size, void* d_ws, size_t ws_size,
                              hipStream_t stream)
{
    const float* x      = (const float*)d_in[0];
    const int*   w_q4   = (const int*)  d_in[1];
    const float* w_norm = (const float*)d_in[2];
    const float* lora_a = (const float*)d_in[3];
    const float* lora_b = (const float*)d_in[4];
    const float* ln_g   = (const float*)d_in[5];
    const float* ln_b   = (const float*)d_in[6];
    float* out = (float*)d_out;

    char* ws = (char*)d_ws;
    size_t off = 0;
    auto alloc = [&](size_t bytes) -> void* {
        void* p = ws + off;
        off += (bytes + 255) & ~(size_t)255;
        return p;
    };
    _Float16* wbuf = (_Float16*)alloc((size_t)CD * CD * sizeof(_Float16));   // 2 MB
    _Float16* fA   = (_Float16*)alloc((size_t)NB * CD * sizeof(_Float16));   // 8 MB
    _Float16* fB   = (_Float16*)alloc((size_t)NB * CD * sizeof(_Float16));   // 8 MB
    _Float16* fC   = (_Float16*)alloc((size_t)NB * CD * sizeof(_Float16));   // 8 MB
    float*    R0   = (float*)   alloc((size_t)NB * CD * sizeof(float));      // 16 MB
    float*    R1   = (float*)   alloc((size_t)NB * CD * sizeof(float));      // 16 MB

    const dim3 gGrid(CD / 128, NB / 128);   // (8, 32)
    const int  dqBlocks = (CD * CD / 2 + 255) / 256;

    auto run_layer = [&](int l, const _Float16* in16, _Float16* o16, float* o32,
                         const float* resid, int relu) {
        dequant_kernel<<<dqBlocks, 256, 0, stream>>>(
            w_q4 + (size_t)l * NGRP * (GS / 2), w_norm + (size_t)l * NGRP,
            lora_a + (size_t)l * 4 * CD, lora_b + (size_t)l * CD * 4, wbuf);
        gemm_kernel<<<gGrid, 256, 0, stream>>>(in16, wbuf, resid, o32, o16, relu);
    };

    // x -> f16
    cvt_kernel<<<(NB * CD + 255) / 256, 256, 0, stream>>>(x, fA, NB * CD);

    // Block 1 (residual = x), then LN0
    run_layer(0, fA, fB, nullptr, nullptr, 1);
    run_layer(1, fB, fC, nullptr, nullptr, 1);
    run_layer(2, fC, fA, R0, x, 0);
    ln_kernel<<<NB, 256, 0, stream>>>(R0, fA, ln_g, ln_b);

    // Block 2 (residual = R0)
    run_layer(3, fA, fB, nullptr, nullptr, 1);
    run_layer(4, fB, fC, nullptr, nullptr, 1);
    run_layer(5, fC, fA, R1, R0, 0);

    // Block 3 (residual = R1), then LN1
    run_layer(6, fA, fB, nullptr, nullptr, 1);
    run_layer(7, fB, fC, nullptr, nullptr, 1);
    run_layer(8, fC, fA, R0, R1, 0);
    ln_kernel<<<NB, 256, 0, stream>>>(R0, fA, ln_g + CD, ln_b + CD);

    // Block 4 (residual = R0) -> d_out
    run_layer(9,  fA, fB, nullptr, nullptr, 1);
    run_layer(10, fB, fC, nullptr, nullptr, 1);
    run_layer(11, fC, fB, out, R0, 0);   // fB as f16 scratch (unused afterwards)
}
